// SuperChessNetwork_62809601736866
// MI455X (gfx1250) — matrix operand order repaired and checked by reading, not compiled, not run on hardware
//
#include <hip/hip_runtime.h>
#include <stddef.h>

// ---------------------------------------------------------------------------
// SuperChessNetwork forward for MI455X (gfx1250), wave32 + WMMA bf16.
// All GEMMs use V_WMMA_F32_16X16X32_BF16 (bf16 in, fp32 accum).
// ---------------------------------------------------------------------------

#define Bn    32
#define Kn    8192
#define TOPKn 1000
#define Fn    1024
#define AEDn  128
#define En    1153          // F + AED + 1
#define ADIMn 4672
#define PFW   1152          // F + AED (policy input width)

typedef __attribute__((ext_vector_type(16))) __bf16 v16bf;
typedef __attribute__((ext_vector_type(8)))  float  v8f;

union FragBF { v16bf v; uint4 u[2]; };   // 32 bytes: 16 bf16 = one WMMA operand/lane

__device__ __forceinline__ float gelu_exact(float x) {
  return 0.5f * x * (1.0f + erff(x * 0.70710678118654752f));
}

// ---------------------------------------------------------------------------
// Elementwise fp32 -> bf16 row copy (strided), used for features / memory[:, :F]
// ---------------------------------------------------------------------------
__global__ void cvt_rows_kernel(const float* __restrict__ src, int srcStride,
                                __bf16* __restrict__ dst, int dstStride,
                                int rows, int cols) {
  size_t idx = (size_t)blockIdx.x * blockDim.x + threadIdx.x;
  size_t total = (size_t)rows * cols;
  if (idx >= total) return;
  int r = (int)(idx / cols), c = (int)(idx % cols);
  dst[(size_t)r * dstStride + c] = (__bf16)src[(size_t)r * srcStride + c];
}

// ---------------------------------------------------------------------------
// W (I x O, fp32, row-major) -> Wt (O x I, bf16) so B-fragments are contiguous.
// block (32,8), grid (O/32, I/32)
// ---------------------------------------------------------------------------
__global__ __launch_bounds__(256) void transpose_cvt_kernel(
    const float* __restrict__ W, int I, int O, __bf16* __restrict__ Wt) {
  __shared__ float tile[32][33];
  int o0 = blockIdx.x * 32, i0 = blockIdx.y * 32;
  int tx = threadIdx.x, ty = threadIdx.y;
#pragma unroll
  for (int j = 0; j < 32; j += 8)
    tile[ty + j][tx] = W[(size_t)(i0 + ty + j) * O + (o0 + tx)];
  __syncthreads();
#pragma unroll
  for (int j = 0; j < 32; j += 8)
    Wt[(size_t)(o0 + ty + j) * I + (i0 + tx)] = (__bf16)tile[tx][ty + j];
}

// ---------------------------------------------------------------------------
// Generic bf16 WMMA GEMM:  C(MxN,f32) = A(MxK,bf16) * Bt(NxK,bf16)^T [+bias][gelu]
// One wave computes a 32x64 tile: 2 M-tiles x 4 N-tiles, 8 wmma per K-step.
// Requires M%32==0, N%64==0, K%32==0 (true for all shapes here).
// ---------------------------------------------------------------------------
__global__ __launch_bounds__(256) void gemm_bf16_wmma(
    const __bf16* __restrict__ A, int lda,
    const __bf16* __restrict__ Bt, int ldb,
    float* __restrict__ C, int ldc,
    __bf16* __restrict__ Cbf, int ldcb,
    const float* __restrict__ bias,
    int M, int N, int Kd, int act) {
  const int waveId = (int)((blockIdx.x * blockDim.x + threadIdx.x) >> 5);
  const int tilesM = M >> 5;
  const int tilesN = N >> 6;
  if (waveId >= tilesM * tilesN) return;
  const int mBase = (waveId % tilesM) << 5;
  const int nBase = (waveId / tilesM) << 6;

  const int lane = threadIdx.x & 31;
  const int half = lane >> 4;
  const int l16  = lane & 15;

  const __bf16* a0p = A + (size_t)(mBase + l16) * lda;
  const __bf16* a1p = A + (size_t)(mBase + 16 + l16) * lda;
  const __bf16* b0p = Bt + (size_t)(nBase +  0 + l16) * ldb + 16 * half;
  const __bf16* b1p = Bt + (size_t)(nBase + 16 + l16) * ldb + 16 * half;
  const __bf16* b2p = Bt + (size_t)(nBase + 32 + l16) * ldb + 16 * half;
  const __bf16* b3p = Bt + (size_t)(nBase + 48 + l16) * ldb + 16 * half;

  const v8f zero = {0.f, 0.f, 0.f, 0.f, 0.f, 0.f, 0.f, 0.f};
  v8f acc[2][4];
#pragma unroll
  for (int i = 0; i < 2; ++i)
#pragma unroll
    for (int j = 0; j < 4; ++j) acc[i][j] = zero;

  for (int k = 0; k < Kd; k += 32) {
    __builtin_prefetch(a0p + k + 256, 0, 1);   // global_prefetch_b8
    __builtin_prefetch(b0p + k + 256, 0, 1);
    FragBF a0, a1, b0, b1, b2, b3;
    // A k order used here (contiguous): lanes 0-15 hold K 0..15, lanes 16-31 hold K 16..31
    a0.u[0] = *(const uint4*)(a0p + k + 16 * half);
    a0.u[1] = *(const uint4*)(a0p + k + 16 * half + 8);
    a1.u[0] = *(const uint4*)(a1p + k + 16 * half);
    a1.u[1] = *(const uint4*)(a1p + k + 16 * half + 8);
    // B uses the same contiguous k order as A: lanes 0-15 hold K 0..15, lanes 16-31 hold K 16..31 (col = lane%16)
    b0.u[0] = *(const uint4*)(b0p + k);
    b0.u[1] = *(const uint4*)(b0p + k + 8);
    b1.u[0] = *(const uint4*)(b1p + k);
    b1.u[1] = *(const uint4*)(b1p + k + 8);
    b2.u[0] = *(const uint4*)(b2p + k);
    b2.u[1] = *(const uint4*)(b2p + k + 8);
    b3.u[0] = *(const uint4*)(b3p + k);
    b3.u[1] = *(const uint4*)(b3p + k + 8);

    acc[0][0] = __builtin_amdgcn_wmma_f32_16x16x32_bf16(false, a0.v, false, b0.v, (short)0, acc[0][0], false, false);
    acc[0][1] = __builtin_amdgcn_wmma_f32_16x16x32_bf16(false, a0.v, false, b1.v, (short)0, acc[0][1], false, false);
    acc[0][2] = __builtin_amdgcn_wmma_f32_16x16x32_bf16(false, a0.v, false, b2.v, (short)0, acc[0][2], false, false);
    acc[0][3] = __builtin_amdgcn_wmma_f32_16x16x32_bf16(false, a0.v, false, b3.v, (short)0, acc[0][3], false, false);
    acc[1][0] = __builtin_amdgcn_wmma_f32_16x16x32_bf16(false, a1.v, false, b0.v, (short)0, acc[1][0], false, false);
    acc[1][1] = __builtin_amdgcn_wmma_f32_16x16x32_bf16(false, a1.v, false, b1.v, (short)0, acc[1][1], false, false);
    acc[1][2] = __builtin_amdgcn_wmma_f32_16x16x32_bf16(false, a1.v, false, b2.v, (short)0, acc[1][2], false, false);
    acc[1][3] = __builtin_amdgcn_wmma_f32_16x16x32_bf16(false, a1.v, false, b3.v, (short)0, acc[1][3], false, false);
  }

  // D layout: vgpr r, lane -> row = r + 8*half, col = lane%16
#pragma unroll
  for (int mt = 0; mt < 2; ++mt)
#pragma unroll
    for (int t = 0; t < 4; ++t)
#pragma unroll
      for (int r = 0; r < 8; ++r) {
        int row = mBase + mt * 16 + half * 8 + r;
        int col = nBase + t * 16 + l16;
        float v = acc[mt][t][r];
        if (bias) v += bias[col];
        if (act == 1) v = gelu_exact(v);
        if (C)   C[(size_t)row * ldc + col] = v;
        if (Cbf) Cbf[(size_t)row * ldcb + col] = (__bf16)v;
      }
}

// ---------------------------------------------------------------------------
// scores[b][k] = sum_h relu(q1b[b][h] + K1[k][h]) * W2s[h]   (H = 512)
// One wave per (b, 64-k chunk); q1b row & W2s cached in registers (16 f/lane).
// (b2s is a scalar shift -> irrelevant to softmax/top-k, omitted.)
// ---------------------------------------------------------------------------
__device__ __forceinline__ float relu_dot4(float4 r, float4 q, float4 w) {
  return fmaxf(r.x + q.x, 0.f) * w.x + fmaxf(r.y + q.y, 0.f) * w.y +
         fmaxf(r.z + q.z, 0.f) * w.z + fmaxf(r.w + q.w, 0.f) * w.w;
}

__global__ __launch_bounds__(256) void scores_relu_dot_kernel(
    const float* __restrict__ K1, const float* __restrict__ q1b,
    const float* __restrict__ W2s, float* __restrict__ scores) {
  const int wave = (int)((blockIdx.x * blockDim.x + threadIdx.x) >> 5);
  const int lane = threadIdx.x & 31;
  const int b = wave >> 7;                 // 128 chunks of 64 k per batch row
  const int kStart = (wave & 127) << 6;
  if (b >= Bn) return;

  const float4* q4 = (const float4*)(q1b + (size_t)b * 512 + lane * 16);
  const float4* w4 = (const float4*)(W2s + lane * 16);
  float4 q0 = q4[0], q1 = q4[1], q2 = q4[2], q3 = q4[3];
  float4 w0 = w4[0], w1 = w4[1], w2 = w4[2], w3 = w4[3];

  for (int kk = 0; kk < 64; ++kk) {
    int k = kStart + kk;
    const float4* r4 = (const float4*)(K1 + (size_t)k * 512 + lane * 16);
    float s = relu_dot4(r4[0], q0, w0) + relu_dot4(r4[1], q1, w1) +
              relu_dot4(r4[2], q2, w2) + relu_dot4(r4[3], q3, w3);
#pragma unroll
    for (int m = 16; m > 0; m >>= 1) s += __shfl_xor(s, m, 32);
    if (lane == 0) scores[(size_t)b * Kn + k] = s;
  }
}

// ---------------------------------------------------------------------------
// Deterministic top-1000 *set* per row via 32-step binary select over
// order-preserving uint keys; ties broken by smallest index. One block per row.
// ---------------------------------------------------------------------------
__global__ __launch_bounds__(256) void topk_kernel(
    const float* __restrict__ scores, int* __restrict__ topk) {
  const int b = blockIdx.x, tid = threadIdx.x;
  __shared__ unsigned keys[Kn];
  __shared__ int sc[256];
  __shared__ unsigned loS, hiS;
  __shared__ int cntS;

  for (int i = tid; i < Kn; i += 256) {
    unsigned bits = __float_as_uint(scores[(size_t)b * Kn + i]);
    keys[i] = bits ^ ((bits & 0x80000000u) ? 0xFFFFFFFFu : 0x80000000u);
  }
  if (tid == 0) { loS = 0u; hiS = 0xFFFFFFFFu; }
  __syncthreads();

  for (int iter = 0; iter < 32; ++iter) {
    unsigned l = loS, h = hiS;
    unsigned mid = l + (unsigned)((((unsigned long long)h - l) + 1ull) >> 1);
    if (tid == 0) cntS = 0;
    __syncthreads();
    if (l < h) {
      int c = 0;
      for (int i = tid; i < Kn; i += 256) c += (keys[i] >= mid) ? 1 : 0;
      if (c) atomicAdd(&cntS, c);
    }
    __syncthreads();
    if (tid == 0 && l < h) { if (cntS >= TOPKn) loS = mid; else hiS = mid - 1; }
    __syncthreads();
  }
  const unsigned T = loS;

  const int per = Kn / 256;                 // contiguous range per thread (32)
  const int base = tid * per;
  int cg = 0, ce = 0;
  for (int i = 0; i < per; ++i) {
    unsigned k = keys[base + i];
    cg += (k > T) ? 1 : 0;
    ce += (k == T) ? 1 : 0;
  }
  // inclusive scan of cg
  sc[tid] = cg; __syncthreads();
  for (int off = 1; off < 256; off <<= 1) {
    int v = (tid >= off) ? sc[tid - off] : 0;
    __syncthreads(); sc[tid] += v; __syncthreads();
  }
  int exg = sc[tid] - cg;
  int totalG = sc[255];
  __syncthreads();
  // inclusive scan of ce
  sc[tid] = ce; __syncthreads();
  for (int off = 1; off < 256; off <<= 1) {
    int v = (tid >= off) ? sc[tid - off] : 0;
    __syncthreads(); sc[tid] += v; __syncthreads();
  }
  int exe = sc[tid] - ce;
  __syncthreads();

  const int need = TOPKn - totalG;
  int* out = topk + (size_t)b * TOPKn;
  int pg = exg, pe = exe;
  for (int i = 0; i < per; ++i) {
    unsigned k = keys[base + i];
    if (k > T) { out[pg++] = base + i; }
    else if (k == T) { if (pe < need) out[totalG + pe] = base + i; ++pe; }
  }
}

// ---------------------------------------------------------------------------
// Stage 2 (one block per batch row):
//   s2[j]   = sum_h gelu(q2b[b][h] + K2all[idx[j]][h]) * W2e[h] + b2e
//   w2      = softmax(s2)
//   wa[a]   = sum_j trunc(memory[idx[j], F+a]) * w2[j]   (also -> pf bf16 tail)
//   wr      = sum_j memory[idx[j], E-1] * w2[j]
// ---------------------------------------------------------------------------
__global__ __launch_bounds__(256) void stage2_kernel(
    const float* __restrict__ memory, const int* __restrict__ topk,
    const float* __restrict__ K2all, const float* __restrict__ q2b,
    const float* __restrict__ W2e, const float* __restrict__ b2e,
    float* __restrict__ out_wa, float* __restrict__ out_wr,
    __bf16* __restrict__ pf) {
  const int b = blockIdx.x, tid = threadIdx.x;
  __shared__ float q2s[256], w2es[256], s2s[TOPKn], red[256];
  __shared__ int idxs[TOPKn];

  if (tid < 256) { q2s[tid] = q2b[(size_t)b * 256 + tid]; w2es[tid] = W2e[tid]; }
  for (int j = tid; j < TOPKn; j += 256) idxs[j] = topk[(size_t)b * TOPKn + j];
  __syncthreads();

  const int wv = tid >> 5, lane = tid & 31;
  const float bias2 = b2e[0];
  for (int j = wv; j < TOPKn; j += 8) {
    const float* row = K2all + (size_t)idxs[j] * 256;
    float acc = 0.f;
    for (int h = lane; h < 256; h += 32)
      acc += gelu_exact(q2s[h] + row[h]) * w2es[h];
#pragma unroll
    for (int m = 16; m > 0; m >>= 1) acc += __shfl_xor(acc, m, 32);
    if (lane == 0) s2s[j] = acc + bias2;
  }
  __syncthreads();

  // softmax over 1000 (fixed deterministic trees)
  float mloc = -3.4e38f;
  for (int j = tid; j < TOPKn; j += 256) mloc = fmaxf(mloc, s2s[j]);
  red[tid] = mloc; __syncthreads();
  for (int off = 128; off > 0; off >>= 1) {
    if (tid < off) red[tid] = fmaxf(red[tid], red[tid + off]);
    __syncthreads();
  }
  float mx = red[0]; __syncthreads();
  float ls = 0.f;
  for (int j = tid; j < TOPKn; j += 256) { float e = expf(s2s[j] - mx); s2s[j] = e; ls += e; }
  red[tid] = ls; __syncthreads();
  for (int off = 128; off > 0; off >>= 1) {
    if (tid < off) red[tid] += red[tid + off];
    __syncthreads();
  }
  float inv = 1.f / red[0]; __syncthreads();
  for (int j = tid; j < TOPKn; j += 256) s2s[j] *= inv;
  __syncthreads();

  if (tid < AEDn) {
    float acc = 0.f;
    for (int j = 0; j < TOPKn; ++j)
      acc += truncf(memory[(size_t)idxs[j] * En + Fn + tid]) * s2s[j];
    out_wa[(size_t)b * AEDn + tid] = acc;
    pf[(size_t)b * PFW + Fn + tid] = (__bf16)acc;
  } else if (wv == 7) {
    float acc = 0.f;
    for (int j = lane; j < TOPKn; j += 32)
      acc += memory[(size_t)idxs[j] * En + (En - 1)] * s2s[j];
#pragma unroll
    for (int m = 16; m > 0; m >>= 1) acc += __shfl_xor(acc, m, 32);
    if (lane == 0) out_wr[b] = acc;
  }
}

// ---------------------------------------------------------------------------
// board = tanh(hv.Wv2 + bv2); aw = sigmoid(hm.Wm2 + bm2); vo = aw*wr + (1-aw)*board
// One wave per batch row (hv/hm already have gelu applied by GEMM epilogue).
// ---------------------------------------------------------------------------
__global__ __launch_bounds__(256) void value_head_kernel(
    const float* __restrict__ hv, const float* __restrict__ hm,
    const float* __restrict__ Wv2, const float* __restrict__ bv2,
    const float* __restrict__ Wm2, const float* __restrict__ bm2,
    const float* __restrict__ wr,
    float* __restrict__ out_bv, float* __restrict__ out_vo) {
  const int wave = (int)((blockIdx.x * blockDim.x + threadIdx.x) >> 5);
  const int lane = threadIdx.x & 31;
  if (wave >= Bn) return;
  float sv = 0.f, sm = 0.f;
  for (int h = lane; h < 512; h += 32) {
    sv += hv[(size_t)wave * 512 + h] * Wv2[h];
    sm += hm[(size_t)wave * 512 + h] * Wm2[h];
  }
#pragma unroll
  for (int m = 16; m > 0; m >>= 1) { sv += __shfl_xor(sv, m, 32); sm += __shfl_xor(sm, m, 32); }
  if (lane == 0) {
    float board = tanhf(sv + bv2[0]);
    float aw = 1.f / (1.f + expf(-(sm + bm2[0])));
    out_bv[wave] = board;
    out_vo[wave] = aw * wr[wave] + (1.f - aw) * board;
  }
}

// ---------------------------------------------------------------------------
// Host side
// ---------------------------------------------------------------------------
static void launch_gemm(hipStream_t stream,
                        const __bf16* A, int lda, const __bf16* Bt, int ldb,
                        float* C, int ldc, __bf16* Cbf, int ldcb,
                        const float* bias, int M, int N, int Kd, int act) {
  int waves = (M >> 5) * (N >> 6);
  int blocks = (waves + 7) / 8;
  gemm_bf16_wmma<<<blocks, 256, 0, stream>>>(A, lda, Bt, ldb, C, ldc, Cbf, ldcb,
                                             bias, M, N, Kd, act);
}

extern "C" void kernel_launch(void* const* d_in, const int* in_sizes, int n_in,
                              void* d_out, int out_size, void* d_ws, size_t ws_size,
                              hipStream_t stream) {
  (void)in_sizes; (void)n_in; (void)out_size; (void)ws_size;

  const float* features = (const float*)d_in[0];
  const float* memory_p = (const float*)d_in[1];
  const float* W1s = (const float*)d_in[2];
  const float* b1s = (const float*)d_in[3];
  const float* W2s = (const float*)d_in[4];
  /* b2s (d_in[5]) is a uniform shift: invariant under softmax/top-k */
  const float* W1e = (const float*)d_in[6];
  const float* b1e = (const float*)d_in[7];
  const float* W2e = (const float*)d_in[8];
  const float* b2e = (const float*)d_in[9];
  const float* Wp1 = (const float*)d_in[10];
  const float* bp1 = (const float*)d_in[11];
  const float* Wp2 = (const float*)d_in[12];
  const float* bp2 = (const float*)d_in[13];
  const float* Wv1 = (const float*)d_in[14];
  const float* bv1 = (const float*)d_in[15];
  const float* Wv2 = (const float*)d_in[16];
  const float* bv2 = (const float*)d_in[17];
  const float* Wm1 = (const float*)d_in[18];
  const float* bm1 = (const float*)d_in[19];
  const float* Wm2 = (const float*)d_in[20];
  const float* bm2 = (const float*)d_in[21];

  float* out        = (float*)d_out;
  float* out_logits = out;                       // 32 x 4672
  float* out_bv     = out + (size_t)Bn * ADIMn;  // 32
  float* out_vo     = out_bv + Bn;               // 32
  float* out_wa     = out_vo + Bn;               // 32 x 128
  float* out_wr     = out_wa + (size_t)Bn * AEDn;// 32

  // --- workspace carve (~58 MB, 256B aligned chunks) ---
  char* p = (char*)d_ws;
  auto alloc = [&](size_t bytes) -> char* {
    char* r = p; p += (bytes + 255) & ~(size_t)255; return r;
  };
  __bf16* featmem = (__bf16*)alloc((size_t)Kn * Fn * 2);     // memory[:, :F] bf16
  float*  K1      = (float*) alloc((size_t)Kn * 512 * 4);
  float*  K2all   = (float*) alloc((size_t)Kn * 256 * 4);
  float*  scoresB = (float*) alloc((size_t)Bn * Kn * 4);
  int*    topkB   = (int*)   alloc((size_t)Bn * TOPKn * 4);
  __bf16* pf      = (__bf16*)alloc((size_t)Bn * PFW * 2);    // [features | wa] bf16
  float*  q1b     = (float*) alloc((size_t)Bn * 512 * 4);
  float*  q2b     = (float*) alloc((size_t)Bn * 256 * 4);
  float*  hv      = (float*) alloc((size_t)Bn * 512 * 4);
  float*  hm      = (float*) alloc((size_t)Bn * 512 * 4);
  float*  HpF     = (float*) alloc((size_t)Bn * 768 * 4);
  __bf16* HpB     = (__bf16*)alloc((size_t)Bn * 768 * 2);
  __bf16* Wq1t    = (__bf16*)alloc((size_t)512 * 1024 * 2);
  __bf16* Wk1t    = (__bf16*)alloc((size_t)512 * 1024 * 2);
  __bf16* Wq2t    = (__bf16*)alloc((size_t)256 * 1024 * 2);
  __bf16* Wk2t    = (__bf16*)alloc((size_t)256 * 1024 * 2);
  __bf16* Wp1t    = (__bf16*)alloc((size_t)768 * PFW * 2);
  __bf16* Wp2t    = (__bf16*)alloc((size_t)ADIMn * 768 * 2);
  __bf16* Wv1t    = (__bf16*)alloc((size_t)512 * 1024 * 2);
  __bf16* Wm1t    = (__bf16*)alloc((size_t)512 * 1024 * 2);

  // --- 1. conversions ---
  {
    size_t tot = (size_t)Kn * Fn;
    cvt_rows_kernel<<<(unsigned)((tot + 255) / 256), 256, 0, stream>>>(
        memory_p, En, featmem, Fn, Kn, Fn);
    size_t tf = (size_t)Bn * Fn;
    cvt_rows_kernel<<<(unsigned)((tf + 255) / 256), 256, 0, stream>>>(
        features, Fn, pf, PFW, Bn, Fn);   // features -> pf[:, :1024]
  }

  // --- 2. weight transposes (fp32 IxO -> bf16 OxI) ---
  dim3 tb(32, 8);
  transpose_cvt_kernel<<<dim3(512 / 32, 1024 / 32), tb, 0, stream>>>(W1s,                 1024, 512,  Wq1t);
  transpose_cvt_kernel<<<dim3(512 / 32, 1024 / 32), tb, 0, stream>>>(W1s + 1024 * 512,    1024, 512,  Wk1t);
  transpose_cvt_kernel<<<dim3(256 / 32, 1024 / 32), tb, 0, stream>>>(W1e,                 1024, 256,  Wq2t);
  transpose_cvt_kernel<<<dim3(256 / 32, 1024 / 32), tb, 0, stream>>>(W1e + 1024 * 256,    1024, 256,  Wk2t);
  transpose_cvt_kernel<<<dim3(768 / 32, PFW / 32),  tb, 0, stream>>>(Wp1,                 PFW,  768,  Wp1t);
  transpose_cvt_kernel<<<dim3(ADIMn / 32, 768 / 32),tb, 0, stream>>>(Wp2,                 768,  ADIMn,Wp2t);
  transpose_cvt_kernel<<<dim3(512 / 32, 1024 / 32), tb, 0, stream>>>(Wv1,                 1024, 512,  Wv1t);
  transpose_cvt_kernel<<<dim3(512 / 32, 1024 / 32), tb, 0, stream>>>(Wm1,                 1024, 512,  Wm1t);

  // --- 3. WMMA GEMMs ---
  launch_gemm(stream, featmem, Fn, Wk1t, Fn, K1,    512, nullptr, 0, nullptr, Kn, 512, Fn, 0); // k1
  launch_gemm(stream, featmem, Fn, Wk2t, Fn, K2all, 256, nullptr, 0, nullptr, Kn, 256, Fn, 0); // K2all
  launch_gemm(stream, pf, PFW, Wq1t, Fn, q1b, 512, nullptr, 0, b1s, Bn, 512, Fn, 0);           // q1 + b1s
  launch_gemm(stream, pf, PFW, Wq2t, Fn, q2b, 256, nullptr, 0, b1e, Bn, 256, Fn, 0);           // q2 + b1e
  launch_gemm(stream, pf, PFW, Wv1t, Fn, hv,  512, nullptr, 0, bv1, Bn, 512, Fn, 1);           // gelu(features.Wv1+bv1)
  launch_gemm(stream, pf, PFW, Wm1t, Fn, hm,  512, nullptr, 0, bm1, Bn, 512, Fn, 1);           // gelu(features.Wm1+bm1)

  // --- 4. scoring + top-k + stage 2 ---
  scores_relu_dot_kernel<<<512, 256, 0, stream>>>(K1, q1b, W2s, scoresB);  // 4096 waves
  topk_kernel<<<Bn, 256, 0, stream>>>(scoresB, topkB);
  stage2_kernel<<<Bn, 256, 0, stream>>>(memory_p, topkB, K2all, q2b, W2e, b2e,
                                        out_wa, out_wr, pf);

  // --- 5. policy head (pf now complete) ---
  launch_gemm(stream, pf, PFW, Wp1t, PFW, HpF, 768, HpB, 768, bp1, Bn, 768, PFW, 1); // gelu fused
  launch_gemm(stream, HpB, 768, Wp2t, 768, out_logits, ADIMn, nullptr, 0, bp2, Bn, ADIMn, 768, 0);

  // --- 6. value combine ---
  value_head_kernel<<<(Bn * 32 + 255) / 256, 256, 0, stream>>>(
      hv, hm, Wv2, bv2, Wm2, bm2, out_wr, out_bv, out_vo);
}
